// WaveBlock_74844100100193
// MI455X (gfx1250) — matrix operand order repaired and checked by reading, not compiled, not run on hardware
//
#include <hip/hip_runtime.h>
#include <hip/hip_bf16.h>
#include <math.h>

// ---------------------------------------------------------------------------
// Types for CDNA5 WMMA (wave32): v_wmma_f32_16x16x32_bf16
// ---------------------------------------------------------------------------
typedef __attribute__((ext_vector_type(16))) __bf16 bfrag;   // A or B fragment
typedef __attribute__((ext_vector_type(8)))  __bf16 v8bf;    // 16-byte unit
typedef __attribute__((ext_vector_type(8)))  float  v8f;     // C/D accumulator

static constexpr int  Bb   = 32;
static constexpr int  Cc   = 96;
static constexpr int  Hh   = 96;
static constexpr int  Ww   = 96;
static constexpr long long NPIX = (long long)Bb * Hh * Ww;   // 294912
static constexpr long long SLAB = 96LL * 96 * 96;            // 884736 per-batch

__device__ __forceinline__ float gelu_exact(float x) {
    return 0.5f * x * (1.0f + erff(x * 0.70710678118654752f));
}
__device__ __forceinline__ float silu(float x) {
    return x / (1.0f + expf(-x));
}

// ---------------------------------------------------------------------------
// WMMA GEMM:  out = A[M,KC] * B[KC,N] (+bias, activation, store-map variants)
//   A : bf16 row-major, lda (K contiguous) -> two b128 global loads / fragment
//   Bt: B^T as [N,KC] row-major, DENSE (ldb == KC). The block's column tile
//       (NT*16 x KC) is staged once in LDS (cooperative, coalesced); fragment
//       reads are then two b128 LDS reads. Rows padded to KC+8 elements so
//       the 16-lane column access hits 16 distinct banks.
// NT: compile-time 16-wide column subtiles per wave (N == gridDim.x*NT*16).
// KC: compile-time K (96 or 384) -> fully unrolled k-loop, static LDS.
// ACT: 0 none | 1 exact gelu
// STORE (address map + dtype; all div/mod by compile-time constants):
//   0 F32_STD   : outF[row*ldc+col]
//   1 BF16_STD  : outB[row*ldc+col]
//   2 BF16_TR96 : outB[(row/96)*9216 + col*96 + row%96]   (vector b128 store)
//   3 F32_SWAP  : outF[col*9216 + (row%96)*96 + row/96]
//   4 F32_PIXTR : outF[(row/9216)*SLAB + col*9216 + row%9216] (vector store)
//   5 SPLIT_LIN : col<96 -> outB[b*SLAB + col*9216 + w*96 + h] (bf16, [b,c,w,h])
//                 col>=96 -> outF[row*96 + col-96]            (f32 std)
// Wave computes 16 rows x NT*16 cols; block = 4 waves sharing the LDS B tile.
// Batch via blockIdx.z (A and out only; Bt shared).
// ---------------------------------------------------------------------------
template<int NT, int KC, int ACT, int STORE>
__global__ __launch_bounds__(128)
void gemm_wmma_kernel(const __bf16* __restrict__ A, int lda, long long strideA,
                      const __bf16* __restrict__ Bt,
                      const float* __restrict__ bias,
                      float* __restrict__ outF, __bf16* __restrict__ outB,
                      int ldc, long long strideC, int M)
{
    constexpr int KP = KC + 8;                    // padded LDS row (bank-safe)
    __shared__ __bf16 Blds[NT * 16 * KP];

    const int tid  = threadIdx.x;
    const int col0 = blockIdx.x * (NT * 16);

    // ---- stage B^T column tile into LDS (all 128 threads, coalesced) ----
    constexpr int CH = (NT * 16 * KC) / 8;        // v8bf chunks
#pragma unroll 2
    for (int ch = tid; ch < CH; ch += 128) {
        const int n  = ch / (KC / 8);
        const int k8 = (ch % (KC / 8)) * 8;
        *(v8bf*)&Blds[n * KP + k8] =
            *(const v8bf*)(Bt + (long long)(col0 + n) * KC + k8);
    }
    __syncthreads();

    const int lane = tid & 31;
    const int wave = tid >> 5;
    const int row0 = blockIdx.y * 64 + wave * 16;
    if (row0 >= M) return;                        // M % 64 == 0 in all launches
    const int lr   = lane & 15;
    const int hi   = (lane >> 4) & 1;
    const long long bz = blockIdx.z;

    A += bz * strideA;

    v8f acc[NT];
#pragma unroll
    for (int t = 0; t < NT; ++t)
#pragma unroll
        for (int i = 0; i < 8; ++i) acc[t][i] = 0.0f;

    // A fragment: lanes 0-15 hold K {0..7,16..23}, lanes 16-31 {8..15,24..31}
    const __bf16* Ap = A + (long long)(row0 + lr) * lda + hi * 8;
    // B fragment from LDS: lane = column, lanes 0-15 K {0..7,16..23}, 16-31 {8..15,24..31}
    const __bf16* Bl = &Blds[lr * KP + hi * 8];

#pragma unroll
    for (int kk = 0; kk < KC; kk += 32) {
        bfrag a;
        {
            const v8bf a0 = *(const v8bf*)(Ap + kk);
            const v8bf a1 = *(const v8bf*)(Ap + kk + 16);
#pragma unroll
            for (int i = 0; i < 8; ++i) { a[i] = a0[i]; a[i + 8] = a1[i]; }
        }
#pragma unroll
        for (int nt = 0; nt < NT; ++nt) {
            const __bf16* Bp = Bl + nt * 16 * KP + kk;
            bfrag b;
            const v8bf b0 = *(const v8bf*)Bp;
            const v8bf b1 = *(const v8bf*)(Bp + 16);
#pragma unroll
            for (int i = 0; i < 8; ++i) { b[i] = b0[i]; b[i + 8] = b1[i]; }
            acc[nt] = __builtin_amdgcn_wmma_f32_16x16x32_bf16(
                false, a, false, b, (short)0, acc[nt], false, false);
        }
    }

    if (outF) outF += bz * strideC;
    if (outB) outB += bz * strideC;

    const int rb = row0 + hi * 8;     // 8 consecutive output rows per lane
#pragma unroll
    for (int nt = 0; nt < NT; ++nt) {
        const int col = col0 + nt * 16 + lr;
        const float bv = bias ? bias[col] : 0.0f;
        if (STORE == 2) {
            // rows rb..rb+7 stay inside one 96-block (rb % 8 == 0, 96 % 8 == 0)
            v8bf vv;
#pragma unroll
            for (int r = 0; r < 8; ++r) {
                float v = acc[nt][r] + bv;
                if (ACT) v = gelu_exact(v);
                vv[r] = (__bf16)v;
            }
            const long long addr = (long long)(rb / 96) * 9216
                                 + (long long)col * 96 + (rb % 96);
            *(v8bf*)(outB + addr) = vv;
        } else if (STORE == 4) {
            v8f vv;
#pragma unroll
            for (int r = 0; r < 8; ++r) {
                float v = acc[nt][r] + bv;
                if (ACT) v = gelu_exact(v);
                vv[r] = v;
            }
            const long long addr = (long long)(rb / 9216) * SLAB
                                 + (long long)col * 9216 + (rb % 9216);
            *(v8f*)(outF + addr) = vv;
        } else {
#pragma unroll
            for (int r = 0; r < 8; ++r) {
                float v = acc[nt][r] + bv;
                if (ACT) v = gelu_exact(v);
                const long long orow = rb + r;
                if (STORE == 0) {
                    outF[orow * ldc + col] = v;
                } else if (STORE == 1) {
                    outB[orow * ldc + col] = (__bf16)v;
                } else if (STORE == 3) {
                    outF[(long long)col * 9216 + (orow % 96) * 96 + orow / 96] = v;
                } else { // 5: lin split
                    const long long pb  = orow / 9216;
                    const int rem = (int)(orow % 9216);
                    const int ph = rem / 96, pw = rem % 96;
                    if (col < Cc)
                        outB[pb * SLAB + (long long)col * 9216 + pw * 96 + ph] = (__bf16)v;
                    else
                        outF[orow * Cc + (col - Cc)] = v;
                }
            }
        }
    }
}

// ---------------------------------------------------------------------------
// Elementwise / small kernels
// ---------------------------------------------------------------------------
__global__ void convert_f32_bf16_kernel(const float* __restrict__ src,
                                        __bf16* __restrict__ dst, long long n)
{
    long long i = (long long)blockIdx.x * blockDim.x + threadIdx.x;
    if (i < n) dst[i] = (__bf16)src[i];
}

// Build DCT-II matrix D[n,h] (and transpose) in bf16, N = 96.
__global__ void build_dct_kernel(__bf16* __restrict__ D, __bf16* __restrict__ Dt)
{
    const int n = blockIdx.x;    // 96 blocks
    const int h = threadIdx.x;   // 96 threads
    float v = cosf((float)n * ((h + 0.5f) / 96.0f) * 3.14159265358979323846f)
              * sqrtf(2.0f / 96.0f);
    if (n == 0) v *= 0.70710678118654752f;
    D[n * 96 + h]  = (__bf16)v;
    Dt[h * 96 + n] = (__bf16)v;
}

// Depthwise 3x3 conv on x[B,C,H,W], output bf16 in NHWC layout.
__global__ void dwconv_kernel(const float* __restrict__ x,
                              const float* __restrict__ wgt,
                              const float* __restrict__ bias,
                              __bf16* __restrict__ out_nhwc)
{
    long long idx = (long long)blockIdx.x * blockDim.x + threadIdx.x;
    const long long total = NPIX * Cc;
    if (idx >= total) return;
    const int c = (int)(idx % Cc);
    const int w = (int)((idx / Cc) % Ww);
    const int h = (int)((idx / (Cc * Ww)) % Hh);
    const int b = (int)(idx / ((long long)Cc * Ww * Hh));
    float acc = bias[c];
    const float* xb = x + (((long long)b * Cc + c) * Hh) * Ww;
#pragma unroll
    for (int kh = 0; kh < 3; ++kh) {
        const int ih = h + kh - 1;
        if (ih < 0 || ih >= Hh) continue;
#pragma unroll
        for (int kw = 0; kw < 3; ++kw) {
            const int iw = w + kw - 1;
            if (iw < 0 || iw >= Ww) continue;
            acc += xb[(long long)ih * Ww + iw] * wgt[c * 9 + kh * 3 + kw];
        }
    }
    out_nhwc[idx] = (__bf16)acc;   // idx is already NHWC-ordered (c fastest)
}

// final = [cos(ct) + sin(ct)/(c+1e-8)*(1+0.5*alpha)] * u0   (v0 == u0)
// u0, t, final all share the [b][c][n][m] layout -> fully coalesced.
__global__ void combine_kernel(const __bf16* __restrict__ u0,
                               const float* __restrict__ t,
                               const float* __restrict__ c_ptr,
                               const float* __restrict__ alpha_ptr,
                               __bf16* __restrict__ out)
{
    long long i = (long long)blockIdx.x * blockDim.x + threadIdx.x;
    const long long total = NPIX * Cc;
    if (i >= total) return;
    const float cv = c_ptr[0];
    const float av = alpha_ptr[0];
    const float ct = cv * t[i];
    const float coef = cosf(ct) + (sinf(ct) / (cv + 1e-8f)) * (1.0f + 0.5f * av);
    out[i] = (__bf16)(coef * (float)u0[i]);
}

__device__ __forceinline__ float row96_reduce(float v, float (*red)[96],
                                              float* stat, int tx, int ty)
{
    red[ty][tx] = v;
    __syncthreads();
    if (tx < 32) {
        float s = red[ty][tx] + red[ty][tx + 32] + red[ty][tx + 64];
        for (int o = 16; o > 0; o >>= 1) s += __shfl_down(s, o);
        if (tx == 0) stat[ty] = s;
    }
    __syncthreads();
    return stat[ty];
}

// g = LayerNorm_C(xf; outnorm, eps=1e-5) * silu(z)   (rows contiguous over C)
__global__ void lnmul_kernel(const float* __restrict__ xf,
                             const float* __restrict__ z,
                             const float* __restrict__ w,
                             const float* __restrict__ b,
                             __bf16* __restrict__ g)
{
    __shared__ float red[4][96];
    __shared__ float smean[4], svar[4];
    const int tx = threadIdx.x;                 // channel
    const int ty = threadIdx.y;
    const long long row = (long long)blockIdx.x * 4 + ty;
    const float v = xf[row * Cc + tx];
    const float mu = row96_reduce(v, red, smean, tx, ty) * (1.0f / 96.0f);
    const float dv = v - mu;
    const float var = row96_reduce(dv * dv, red, svar, tx, ty) * (1.0f / 96.0f);
    const float y = dv * rsqrtf(var + 1e-5f) * w[tx] + b[tx];
    const float zv = z[row * Cc + tx];
    g[row * Cc + tx] = (__bf16)(y * silu(zv));
}

// norm1: x1 = x + LN_W(op; eps=1e-6). op stored NHWC, LN along W (stride C).
// Writes x1 f32 (NCHW) and x1 bf16 (NHWC) for the following 1x1-conv GEMM.
__global__ void norm1_kernel(const float* __restrict__ x_nchw,
                             const float* __restrict__ op_nhwc,
                             const float* __restrict__ nw,
                             const float* __restrict__ nb,
                             float* __restrict__ x1f,
                             __bf16* __restrict__ x1b)
{
    __shared__ float red[4][96];
    __shared__ float smean[4], svar[4];
    const int tx = threadIdx.x;                 // w index
    const int ty = threadIdx.y;
    const long long r = (long long)blockIdx.x * 4 + ty;   // (b,c,h)
    const int h = (int)(r % Hh);
    const int c = (int)((r / Hh) % Cc);
    const int b = (int)(r / ((long long)Hh * Cc));
    const long long nh = (((long long)b * Hh + h) * Ww + tx) * Cc + c;
    const float v = op_nhwc[nh];
    const float mu = row96_reduce(v, red, smean, tx, ty) * (1.0f / 96.0f);
    const float dv = v - mu;
    const float var = row96_reduce(dv * dv, red, svar, tx, ty) * (1.0f / 96.0f);
    const float y = dv * rsqrtf(var + 1e-6f) * nw[tx] + nb[tx];
    const long long ni = (((long long)b * Cc + c) * Hh + h) * Ww + tx;
    const float o = x_nchw[ni] + y;
    x1f[ni] = o;
    x1b[nh] = (__bf16)o;
}

// norm2: out = x1 + LN_W(m2; eps=1e-6). m2 stored NHWC; out NCHW f32.
__global__ void norm2_kernel(const float* __restrict__ x1f,
                             const float* __restrict__ m2_nhwc,
                             const float* __restrict__ nw,
                             const float* __restrict__ nb,
                             float* __restrict__ out)
{
    __shared__ float red[4][96];
    __shared__ float smean[4], svar[4];
    const int tx = threadIdx.x;
    const int ty = threadIdx.y;
    const long long r = (long long)blockIdx.x * 4 + ty;
    const int h = (int)(r % Hh);
    const int c = (int)((r / Hh) % Cc);
    const int b = (int)(r / ((long long)Hh * Cc));
    const long long nh = (((long long)b * Hh + h) * Ww + tx) * Cc + c;
    const float v = m2_nhwc[nh];
    const float mu = row96_reduce(v, red, smean, tx, ty) * (1.0f / 96.0f);
    const float dv = v - mu;
    const float var = row96_reduce(dv * dv, red, svar, tx, ty) * (1.0f / 96.0f);
    const float y = dv * rsqrtf(var + 1e-6f) * nw[tx] + nb[tx];
    const long long ni = (((long long)b * Cc + c) * Hh + h) * Ww + tx;
    out[ni] = x1f[ni] + y;
}

// ---------------------------------------------------------------------------
// Orchestration
// ---------------------------------------------------------------------------
extern "C" void kernel_launch(void* const* d_in, const int* in_sizes, int n_in,
                              void* d_out, int out_size, void* d_ws, size_t ws_size,
                              hipStream_t stream)
{
    (void)in_sizes; (void)n_in; (void)out_size; (void)ws_size;
    const float* x         = (const float*)d_in[0];
    const float* freq      = (const float*)d_in[1];
    const float* dwconv_w  = (const float*)d_in[2];
    const float* dwconv_b  = (const float*)d_in[3];
    const float* lin_w     = (const float*)d_in[4];
    const float* lin_b     = (const float*)d_in[5];
    const float* tok_w     = (const float*)d_in[6];
    const float* tok_b     = (const float*)d_in[7];
    const float* c_s       = (const float*)d_in[8];
    const float* alpha_s   = (const float*)d_in[9];
    const float* outnorm_w = (const float*)d_in[10];
    const float* outnorm_b = (const float*)d_in[11];
    const float* outlin_w  = (const float*)d_in[12];
    const float* outlin_b  = (const float*)d_in[13];
    const float* norm1_w   = (const float*)d_in[14];
    const float* norm1_b   = (const float*)d_in[15];
    const float* fc1_w     = (const float*)d_in[16];
    const float* fc1_b     = (const float*)d_in[17];
    const float* fc2_w     = (const float*)d_in[18];
    const float* fc2_b     = (const float*)d_in[19];
    const float* norm2_w   = (const float*)d_in[20];
    const float* norm2_b   = (const float*)d_in[21];
    float* out = (float*)d_out;

    // --- workspace carve-up (256B aligned) ---
    char* ws = (char*)d_ws;
    size_t off = 0;
    auto carve = [&](size_t bytes) -> char* {
        char* p = ws + off;
        off += (bytes + 255) & ~(size_t)255;
        return p;
    };
    const size_t BF_POOL = (size_t)NPIX * Cc * sizeof(__bf16);   // 56.6 MB
    const size_t F_POOL  = (size_t)NPIX * Cc * sizeof(float);    // 113 MB

    __bf16* Dm   = (__bf16*)carve(96 * 96 * 2);
    __bf16* Dt   = (__bf16*)carve(96 * 96 * 2);
    __bf16* wlin = (__bf16*)carve(192 * 96 * 2);
    __bf16* wtok = (__bf16*)carve(96 * 96 * 2);
    __bf16* wout = (__bf16*)carve(96 * 96 * 2);
    __bf16* wfc1 = (__bf16*)carve(384 * 96 * 2);
    __bf16* wfc2 = (__bf16*)carve(96 * 384 * 2);
    __bf16* P0 = (__bf16*)carve(BF_POOL);   // d / freq_bf16 / g
    __bf16* P1 = (__bf16*)carve(BF_POOL);   // xx_t / final_t / x1_bf16
    __bf16* P2 = (__bf16*)carve(BF_POOL);   // u1_t / xw_t
    __bf16* P3 = (__bf16*)carve(BF_POOL);   // u0_t
    float*  Q0 = (float*)carve(F_POOL);     // z  -> m2
    float*  Q1 = (float*)carve(F_POOL);     // t_t -> xf
    float*  Q2 = (float*)carve(F_POOL);     // op
    float*  Q3 = (float*)carve(F_POOL);     // x1 f32 (NCHW)
    __bf16* R  = (__bf16*)carve((size_t)NPIX * 384 * 2);  // m1 (gelu, bf16)

    const int T = 256;
    auto ceil_div = [](long long a, long long b) { return (int)((a + b - 1) / b); };

    // 1) DCT matrix + weight conversions to bf16
    build_dct_kernel<<<96, 96, 0, stream>>>(Dm, Dt);
    convert_f32_bf16_kernel<<<ceil_div(192 * 96, T), T, 0, stream>>>(lin_w,  wlin, 192 * 96);
    convert_f32_bf16_kernel<<<ceil_div(96 * 96, T),  T, 0, stream>>>(tok_w,  wtok, 96 * 96);
    convert_f32_bf16_kernel<<<ceil_div(96 * 96, T),  T, 0, stream>>>(outlin_w, wout, 96 * 96);
    convert_f32_bf16_kernel<<<ceil_div(384 * 96, T), T, 0, stream>>>(fc1_w,  wfc1, 384 * 96);
    convert_f32_bf16_kernel<<<ceil_div(96 * 384, T), T, 0, stream>>>(fc2_w,  wfc2, 96 * 384);

    // 2) depthwise conv -> NHWC bf16 (P0)
    dwconv_kernel<<<ceil_div(NPIX * Cc, T), T, 0, stream>>>(x, dwconv_w, dwconv_b, P0);

    // 3) lin: split -> xx_t bf16 [b][c][w][h] (P1) | z f32 std (Q0)
    gemm_wmma_kernel<4, 96, 0, 5><<<dim3(3, 4608, 1), 128, 0, stream>>>(
        P0, 96, 0, wlin, lin_b, Q0, P1, 96, 0, (int)NPIX);

    // 4) tok: freq -> bf16 (P0), gelu(freq @ tok_w.T + b) -> t_t [b][c][n][m] (Q1)
    convert_f32_bf16_kernel<<<ceil_div(NPIX * Cc, T), T, 0, stream>>>(freq, P0, NPIX * Cc);
    gemm_wmma_kernel<6, 96, 1, 4><<<dim3(1, 4608, 1), 128, 0, stream>>>(
        P0, 96, 0, wtok, tok_b, Q1, nullptr, 96, 0, (int)NPIX);

    // 5) DCT-H: rows (c,w), contract h: u1_t [b][c][n][w] (P2)
    gemm_wmma_kernel<6, 96, 0, 2><<<dim3(1, 144, 32), 128, 0, stream>>>(
        P1, 96, SLAB, Dm, nullptr, nullptr, P2, 96, SLAB, 9216);

    // 6) DCT-W: rows (c,n), contract w: u0_t [b][c][n][m] (P3)
    gemm_wmma_kernel<6, 96, 0, 1><<<dim3(1, 144, 32), 128, 0, stream>>>(
        P2, 96, SLAB, Dm, nullptr, nullptr, P3, 96, SLAB, 9216);

    // 7) frequency-domain modulation -> final_t [b][c][n][m] (P1)
    combine_kernel<<<ceil_div(NPIX * Cc, T), T, 0, stream>>>(P3, Q1, c_s, alpha_s, P1);

    // 8) IDCT-W: rows (c,n), contract m: xw_t [b][c][w][n] (P2)
    gemm_wmma_kernel<6, 96, 0, 2><<<dim3(1, 144, 32), 128, 0, stream>>>(
        P1, 96, SLAB, Dt, nullptr, nullptr, P2, 96, SLAB, 9216);

    // 9) IDCT-H: rows (c,w), contract n: xf f32 std [b,h,w,c] (Q1)
    gemm_wmma_kernel<6, 96, 0, 3><<<dim3(1, 144, 32), 128, 0, stream>>>(
        P2, 96, SLAB, Dt, nullptr, Q1, nullptr, 96, SLAB, 9216);

    // 10) g = LN_C(xf)*silu(z) -> bf16 (P0)
    lnmul_kernel<<<(unsigned)(NPIX / 4), dim3(96, 4), 0, stream>>>(
        Q1, Q0, outnorm_w, outnorm_b, P0);

    // 11) outlin: op f32 (Q2, NHWC)
    gemm_wmma_kernel<6, 96, 0, 0><<<dim3(1, 4608, 1), 128, 0, stream>>>(
        P0, 96, 0, wout, outlin_b, Q2, nullptr, 96, 0, (int)NPIX);

    // 12) norm1 residual: x1 = x + LN_W(op) -> Q3 (f32 NCHW) + P1 (bf16 NHWC)
    norm1_kernel<<<(unsigned)((long long)Bb * Cc * Hh / 4), dim3(96, 4), 0, stream>>>(
        x, Q2, norm1_w, norm1_b, Q3, P1);

    // 13) fc1: gelu(x1 @ fc1_w.T + b) -> m1 bf16 (R)
    gemm_wmma_kernel<4, 96, 1, 1><<<dim3(6, 4608, 1), 128, 0, stream>>>(
        P1, 96, 0, wfc1, fc1_b, nullptr, R, 384, 0, (int)NPIX);

    // 14) fc2: m2 f32 (Q0, NHWC)
    gemm_wmma_kernel<6, 384, 0, 0><<<dim3(1, 4608, 1), 128, 0, stream>>>(
        R, 384, 0, wfc2, fc2_b, Q0, nullptr, 96, 0, (int)NPIX);

    // 15) norm2 residual: out = x1 + LN_W(m2) -> d_out (f32 NCHW)
    norm2_kernel<<<(unsigned)((long long)Bb * Cc * Hh / 4), dim3(96, 4), 0, stream>>>(
        Q3, Q0, norm2_w, norm2_b, out);
}
